// TDID_36679020707933
// MI455X (gfx1250) — matrix operand order repaired and checked by reading, not compiled, not run on hardware
//
#include <hip/hip_runtime.h>
#include <hip/hip_bf16.h>
#include <math.h>

typedef __attribute__((ext_vector_type(16))) _Float16 v16h;
typedef __attribute__((ext_vector_type(8)))  _Float16 v8h;
typedef __attribute__((ext_vector_type(8)))  float    v8f;

#define HW      64
#define NPIX    4096
#define BATCH   4
#define NTGT    2
#define NCAND   36864      /* 4096*9 */
#define SORTN   65536
#define PRE_NMS 6000
#define POST_NMS 300
#define NEGV    (-1000000000.0f)
#define BSTR    40         /* LDS row stride in halves (padded, keeps b128 reads aligned) */

/* anchors from _generate_anchors(16,(.5,1,2),(8,16,32)) evaluated exactly */
__constant__ float ANCH[9][4] = {
    { -84.f,  -40.f,  99.f,  55.f}, {-176.f,  -88.f, 191.f, 103.f}, {-360.f, -184.f, 375.f, 199.f},
    { -56.f,  -56.f,  71.f,  71.f}, {-120.f, -120.f, 135.f, 135.f}, {-248.f, -248.f, 263.f, 263.f},
    { -36.f,  -80.f,  51.f,  95.f}, { -80.f, -168.f,  95.f, 183.f}, {-168.f, -344.f, 183.f, 359.f}};

/* -------- pooled[b][t][c] = max_{5x5} tgt_feats -------- */
__global__ void pool_kernel(const float* __restrict__ tgt, float* __restrict__ pooled) {
    int t = blockIdx.x * 256 + threadIdx.x;
    if (t >= BATCH * NTGT * 512) return;
    const float* s = tgt + (size_t)t * 25;
    float m = s[0];
    #pragma unroll
    for (int i = 1; i < 25; ++i) m = fmaxf(m, s[i]);
    pooled[t] = m;
}

/* -------- W(512,Cin,3,3) f32 -> Wh(9,Cin,512) f16 -------- */
__global__ void wconv_cvt(const float* __restrict__ W, _Float16* __restrict__ outp, int Cin) {
    size_t t = (size_t)blockIdx.x * 256 + threadIdx.x;
    size_t n = (size_t)512 * Cin * 9;
    if (t >= n) return;
    int cout = (int)(t % 512);
    size_t rem = t / 512;
    int cin = (int)(rem % Cin);
    int r   = (int)(rem / Cin);
    outp[t] = (_Float16)W[((size_t)cout * Cin + cin) * 9 + r];
}

/* -------- combined (Wscore;Wbbox) -> f16 (54,512) -------- */
__global__ void wsb_cvt(const float* __restrict__ Ws, const float* __restrict__ Wb,
                        _Float16* __restrict__ outp) {
    int t = blockIdx.x * 256 + threadIdx.x;
    if (t >= 54 * 512) return;
    int o = t / 512, c = t % 512;
    outp[t] = (_Float16)(o < 18 ? Ws[o * 512 + c] : Wb[(o - 18) * 512 + c]);
}

/* -------- copy im_feats into middle 512 channels of feat (f16) -------- */
__global__ void imcopy_kernel(const float* __restrict__ im, _Float16* __restrict__ feat) {
    size_t t = (size_t)blockIdx.x * 256 + threadIdx.x;
    if (t >= (size_t)BATCH * 512 * NPIX) return;
    int p = (int)(t & (NPIX - 1));
    size_t rem = t >> 12;
    int c = (int)(rem & 511);
    int b = (int)(rem >> 9);
    feat[((size_t)(b * 1536 + 512 + c)) * NPIX + p] = (_Float16)im[t];
}

__global__ void zkey_kernel(unsigned long long* __restrict__ skey) {
    size_t t = (size_t)blockIdx.x * 256 + threadIdx.x;
    if (t < (size_t)BATCH * SORTN) skey[t] = 0ull;
}

/* -------- implicit-GEMM 3x3 SAME conv via v_wmma_f32_16x16x32_f16 --------
   mode 0: in = im*pooled (cin -> t=cin/512,c=cin%512), Cin=1024
   mode 1: in = im-pooled,                             Cin=1024
   mode 2: in = feat f16 (1536 ch),                    Cin=1536
   WG: 256 thr (8 waves), tile 128 couts x 64 pixels (= one image row y). */
__global__ __launch_bounds__(256)
void conv3_wmma(const float* __restrict__ im,
                const _Float16* __restrict__ fin,
                const float* __restrict__ pooled,
                const _Float16* __restrict__ Wh,     /* (9,Cin,512) f16 */
                const float* __restrict__ bias,
                _Float16* __restrict__ outp,
                int mode, int Cin, int obstride, int cbase)
{
    __shared__ __attribute__((aligned(16))) _Float16 Alds[128 * BSTR];
    __shared__ __attribute__((aligned(16))) _Float16 Blds[64 * BSTR];

    const int tid  = threadIdx.x;
    const int lane = tid & 31;
    const int wave = tid >> 5;
    const int y     = blockIdx.x;        /* 0..63 : one image row per WG */
    const int cout0 = blockIdx.y << 7;   /* 0,128,256,384 */
    const int b     = blockIdx.z;

    const int kl  = tid >> 3;            /* 0..31 staging k row */
    const int sub = tid & 7;             /* 0..7  staging chunk */
    const int g   = lane >> 4;           /* lane group */
    const int hn  = lane & 15;

    v8f zero = {};
    v8f acc[4];
    #pragma unroll
    for (int j = 0; j < 4; ++j) acc[j] = zero;

    for (int r = 0; r < 9; ++r) {
        const int dy = r / 3 - 1;
        const int dx = r % 3 - 1;
        const int yy = y + dy;
        const bool rowok = (yy >= 0) && (yy < HW);
        const int ycl = rowok ? yy : 0;

        for (int cb0 = 0; cb0 < Cin; cb0 += 32) {
            /* ---- stage A tile: 128 couts x 32 k, LDS layout [cout][k] ---- */
            {
                const _Float16* asrc =
                    Wh + (size_t)(r * Cin + cb0 + kl) * 512 + cout0 + sub * 16;
                union { uint4 u; _Float16 h[8]; } t0, t1;
                t0.u = *(const uint4*)asrc;
                t1.u = *(const uint4*)(asrc + 8);
                _Float16* adst = &Alds[(sub * 16) * BSTR + kl];
                #pragma unroll
                for (int i = 0; i < 8; ++i) adst[i * BSTR] = t0.h[i];
                #pragma unroll
                for (int i = 0; i < 8; ++i) adst[(8 + i) * BSTR] = t1.h[i];
            }
            /* ---- stage B tile: 32 k x 64 pixels, LDS layout [pixel][k] ---- */
            {
                const int cin = cb0 + kl;
                _Float16 vals[8];
                if (mode == 2) {
                    const _Float16* bsrc =
                        fin + (((size_t)b * 1536 + cin) * HW + ycl) * HW;
                    #pragma unroll
                    for (int i = 0; i < 8; ++i) {
                        int xx = sub * 8 + i + dx;
                        bool ok = rowok && (xx >= 0) && (xx < HW);
                        vals[i] = ok ? bsrc[xx] : (_Float16)0.f;
                    }
                } else {
                    const int tt = cin >> 9;
                    const int c  = cin & 511;
                    const float pv = pooled[((size_t)b * NTGT + tt) * 512 + c];
                    const float* bsrc =
                        im + (((size_t)b * 512 + c) * HW + ycl) * HW;
                    #pragma unroll
                    for (int i = 0; i < 8; ++i) {
                        int xx = sub * 8 + i + dx;
                        bool ok = rowok && (xx >= 0) && (xx < HW);
                        float v = ok ? bsrc[xx] : 0.f;
                        float o = (mode == 0) ? v * pv : v - pv;
                        vals[i] = ok ? (_Float16)o : (_Float16)0.f;
                    }
                }
                _Float16* bdst = &Blds[kl];
                #pragma unroll
                for (int i = 0; i < 8; ++i) bdst[(sub * 8 + i) * BSTR] = vals[i];
            }
            __syncthreads();

            /* ---- fragments per ISA lane layouts, 4 WMMA per wave per k-step ---- */
            v16h afrag;
            {
                const _Float16* ap = &Alds[(wave * 16 + hn) * BSTR + g * 8];
                union { v16h v; v8h h[2]; } ua;
                ua.h[0] = *(const v8h*)ap;        /* k = 8g .. 8g+7   */
                ua.h[1] = *(const v8h*)(ap + 16); /* k = 16+8g .. +7  */
                afrag = ua.v;
            }
            #pragma unroll
            for (int jt = 0; jt < 4; ++jt) {
                const _Float16* bp = &Blds[(jt * 16 + hn) * BSTR + g * 8];
                union { v16h v; v8h h[2]; } ub;
                ub.h[0] = *(const v8h*)bp;        /* k = 8g .. 8g+7   */
                ub.h[1] = *(const v8h*)(bp + 16); /* k = 16+8g .. +7  */
                acc[jt] = __builtin_amdgcn_wmma_f32_16x16x32_f16(
                    false, afrag, false, ub.v, (short)0, acc[jt], false, false);
            }
            __syncthreads();
        }
    }

    /* ---- epilogue: D layout M = 8g+v, N = hn ; bias + relu, store f16 ---- */
    #pragma unroll
    for (int jt = 0; jt < 4; ++jt) {
        const int x = jt * 16 + hn;
        #pragma unroll
        for (int v = 0; v < 8; ++v) {
            const int co = cout0 + wave * 16 + g * 8 + v;
            float val = acc[jt][v] + bias[co];
            val = val > 0.f ? val : 0.f;
            outp[(size_t)b * obstride + (size_t)(cbase + co) * NPIX + y * HW + x] =
                (_Float16)val;
        }
    }
}

/* -------- 1x1 convs + softmax + anchor decode + sort-key pack -------- */
__global__ __launch_bounds__(256)
void head_kernel(const _Float16* __restrict__ rpn,
                 const _Float16* __restrict__ Wsb,   /* (54,512) f16 */
                 const float* __restrict__ bsc,      /* (18) */
                 const float* __restrict__ bbb,      /* (36) */
                 float* __restrict__ out_prob,       /* (B,18,64,64) */
                 float* __restrict__ out_bbox,       /* (B,36,64,64) */
                 unsigned long long* __restrict__ skey,
                 float* __restrict__ props,          /* (B,NCAND,4) */
                 float* __restrict__ fscore)         /* (B,NCAND)   */
{
    __shared__ _Float16 WL[54 * 512];
    const int tid = threadIdx.x;
    const int b = blockIdx.y;
    const int p = blockIdx.x * 256 + tid;
    for (int i = tid; i < 54 * 512; i += 256) WL[i] = Wsb[i];
    __syncthreads();

    float acc[54];
    #pragma unroll
    for (int o = 0; o < 54; ++o) acc[o] = 0.f;
    const _Float16* rp = rpn + (size_t)b * 512 * NPIX + p;
    for (int c = 0; c < 512; ++c) {
        const float v = (float)rp[(size_t)c * NPIX];
        #pragma unroll
        for (int o = 0; o < 54; ++o) acc[o] += v * (float)WL[o * 512 + c];
    }

    const int yq = p >> 6, xq = p & 63;
    const float sx = (float)(xq * 16), sy = (float)(yq * 16);

    float bb[36];
    #pragma unroll
    for (int o = 0; o < 36; ++o) {
        bb[o] = acc[18 + o] + bbb[o];
        out_bbox[((size_t)b * 36 + o) * NPIX + p] = bb[o];
    }
    #pragma unroll
    for (int a = 0; a < 9; ++a) {
        float c0 = acc[a] + bsc[a];
        float c1 = acc[9 + a] + bsc[9 + a];
        float m  = fmaxf(c0, c1);
        float e0 = expf(c0 - m), e1 = expf(c1 - m);
        float inv = 1.f / (e0 + e1);
        float pbg = e0 * inv, pfg = e1 * inv;
        out_prob[((size_t)b * 18 + a) * NPIX + p]     = pbg;
        out_prob[((size_t)b * 18 + 9 + a) * NPIX + p] = pfg;

        /* decode + clip */
        float ax1 = ANCH[a][0] + sx, ay1 = ANCH[a][1] + sy;
        float ax2 = ANCH[a][2] + sx, ay2 = ANCH[a][3] + sy;
        float w = ax2 - ax1 + 1.f, h = ay2 - ay1 + 1.f;
        float cx = ax1 + 0.5f * w,  cy = ay1 + 0.5f * h;
        float pcx = bb[4 * a] * w + cx, pcy = bb[4 * a + 1] * h + cy;
        float pw = expf(bb[4 * a + 2]) * w, ph = expf(bb[4 * a + 3]) * h;
        float x1 = fminf(fmaxf(pcx - 0.5f * pw, 0.f), 1023.f);
        float y1 = fminf(fmaxf(pcy - 0.5f * ph, 0.f), 1023.f);
        float x2 = fminf(fmaxf(pcx + 0.5f * pw, 0.f), 1023.f);
        float y2 = fminf(fmaxf(pcy + 0.5f * ph, 0.f), 1023.f);

        const unsigned int ci = (unsigned int)(p * 9 + a);
        *(float4*)(props + ((size_t)b * NCAND + ci) * 4) = make_float4(x1, y1, x2, y2);
        fscore[(size_t)b * NCAND + ci] = pfg;
        /* order-preserving key (pfg > 0): high bits = score, low = ~idx (stable ties) */
        unsigned int ub = __float_as_uint(pfg) | 0x80000000u;
        skey[(size_t)b * SORTN + ci] =
            ((unsigned long long)ub << 32) | (unsigned int)(~ci);
    }
}

/* -------- per-batch bitonic sort (descending) of 65536 packed keys -------- */
__global__ __launch_bounds__(1024)
void sort_kernel(unsigned long long* __restrict__ skey) {
    unsigned long long* a = skey + (size_t)blockIdx.x * SORTN;
    const int tid = threadIdx.x;
    for (int ksz = 2; ksz <= SORTN; ksz <<= 1) {
        for (int j = ksz >> 1; j > 0; j >>= 1) {
            for (int i = tid; i < SORTN; i += 1024) {
                int ixj = i ^ j;
                if (ixj > i) {
                    unsigned long long va = a[i], vb = a[ixj];
                    bool up = ((i & ksz) == 0);
                    if (up ? (va < vb) : (va > vb)) { a[i] = vb; a[ixj] = va; }
                }
            }
            __threadfence_block();
            __syncthreads();
            __threadfence_block();
        }
    }
}

/* -------- per-batch greedy NMS, 300 iterations over top-6000 -------- */
__global__ __launch_bounds__(1024)
void nms_kernel(const unsigned long long* __restrict__ skey,
                const float* __restrict__ props,
                const float* __restrict__ fscore,
                float* __restrict__ nbx,   /* (B,5,6000): x1,y1,x2,y2,area */
                float* __restrict__ nsc,   /* (B,6000) sorted scores */
                float* __restrict__ out_rois,
                float* __restrict__ out_ks)
{
    __shared__ float work[PRE_NMS];
    __shared__ float rv[1024];
    __shared__ int   ri[1024];
    __shared__ float bc[5];
    const int tid = threadIdx.x;
    const int b = blockIdx.x;
    const unsigned long long* sk = skey + (size_t)b * SORTN;
    float* X1 = nbx + (size_t)b * PRE_NMS * 5;
    float* Y1 = X1 + PRE_NMS;
    float* X2 = Y1 + PRE_NMS;
    float* Y2 = X2 + PRE_NMS;
    float* AR = Y2 + PRE_NMS;
    float* SC = nsc + (size_t)b * PRE_NMS;

    for (int i = tid; i < PRE_NMS; i += 1024) {
        unsigned long long kv = sk[i];
        unsigned int ci = ~(unsigned int)(kv & 0xFFFFFFFFull);
        float s = fscore[(size_t)b * NCAND + ci];
        const float* pr = props + ((size_t)b * NCAND + ci) * 4;
        float x1 = pr[0], y1 = pr[1], x2 = pr[2], y2 = pr[3];
        X1[i] = x1; Y1[i] = y1; X2[i] = x2; Y2[i] = y2;
        AR[i] = (x2 - x1 + 1.f) * (y2 - y1 + 1.f);
        SC[i] = s;
        work[i] = s;
    }
    __threadfence_block();
    __syncthreads();
    __threadfence_block();

    for (int it = 0; it < POST_NMS; ++it) {
        /* argmax(work) with first-occurrence tie-break */
        float bv = -3e38f; int bi = 0x7FFFFFFF;
        for (int i = tid; i < PRE_NMS; i += 1024) {
            float w = work[i];
            if (w > bv || (w == bv && i < bi)) { bv = w; bi = i; }
        }
        rv[tid] = bv; ri[tid] = bi;
        __syncthreads();
        for (int s = 512; s > 0; s >>= 1) {
            if (tid < s) {
                float ov = rv[tid + s]; int oi = ri[tid + s];
                if (ov > rv[tid] || (ov == rv[tid] && oi < ri[tid])) {
                    rv[tid] = ov; ri[tid] = oi;
                }
            }
            __syncthreads();
        }
        if (tid == 0) {
            int j = ri[0];
            bool valid = rv[0] > NEGV * 0.5f;
            bc[0] = X1[j]; bc[1] = Y1[j]; bc[2] = X2[j]; bc[3] = Y2[j]; bc[4] = AR[j];
            float* ro = out_rois + ((size_t)b * POST_NMS + it) * 5;
            ro[0] = (float)b;
            ro[1] = valid ? X1[j] : 0.f;
            ro[2] = valid ? Y1[j] : 0.f;
            ro[3] = valid ? X2[j] : 0.f;
            ro[4] = valid ? Y2[j] : 0.f;
            out_ks[b * POST_NMS + it] = valid ? SC[j] : 0.f;
        }
        __syncthreads();
        float jx1 = bc[0], jy1 = bc[1], jx2 = bc[2], jy2 = bc[3], ja = bc[4];
        for (int i = tid; i < PRE_NMS; i += 1024) {
            float xx1 = fmaxf(jx1, X1[i]);
            float yy1 = fmaxf(jy1, Y1[i]);
            float xx2 = fminf(jx2, X2[i]);
            float yy2 = fminf(jy2, Y2[i]);
            float inter = fmaxf(xx2 - xx1 + 1.f, 0.f) * fmaxf(yy2 - yy1 + 1.f, 0.f);
            float iou = inter / (ja + AR[i] - inter);
            if (iou > 0.7f) work[i] = NEGV;
        }
        __syncthreads();
    }
}

extern "C" void kernel_launch(void* const* d_in, const int* in_sizes, int n_in,
                              void* d_out, int out_size, void* d_ws, size_t ws_size,
                              hipStream_t stream)
{
    const float* im  = (const float*)d_in[0];
    const float* tgt = (const float*)d_in[1];
    const float* Wcc = (const float*)d_in[2];
    const float* bcc = (const float*)d_in[3];
    const float* Wdf = (const float*)d_in[4];
    const float* bdf = (const float*)d_in[5];
    const float* Wc1 = (const float*)d_in[6];
    const float* bc1 = (const float*)d_in[7];
    const float* Wsc = (const float*)d_in[8];
    const float* bsc = (const float*)d_in[9];
    const float* Wbb = (const float*)d_in[10];
    const float* bbb = (const float*)d_in[11];
    float* out = (float*)d_out;

    char* ws = (char*)d_ws;
    size_t off = 0;
    auto carve = [&](size_t bytes) -> void* {
        void* p = (void*)(ws + off);
        off = (off + bytes + 255) & ~(size_t)255;
        return p;
    };
    float*    pooled = (float*)carve(sizeof(float) * BATCH * NTGT * 512);
    _Float16* WccH   = (_Float16*)carve(sizeof(_Float16) * 9 * 1024 * 512);
    _Float16* WdfH   = (_Float16*)carve(sizeof(_Float16) * 9 * 1024 * 512);
    _Float16* Wc1H   = (_Float16*)carve(sizeof(_Float16) * 9 * 1536 * 512);
    _Float16* WsbH   = (_Float16*)carve(sizeof(_Float16) * 54 * 512);
    _Float16* featH  = (_Float16*)carve(sizeof(_Float16) * BATCH * 1536 * NPIX);
    _Float16* rpnH   = (_Float16*)carve(sizeof(_Float16) * BATCH * 512 * NPIX);
    float*    props  = (float*)carve(sizeof(float) * BATCH * NCAND * 4);
    float*    fsc    = (float*)carve(sizeof(float) * BATCH * NCAND);
    unsigned long long* skey = (unsigned long long*)carve(8ull * BATCH * SORTN);
    float*    nbx    = (float*)carve(sizeof(float) * BATCH * PRE_NMS * 5);
    float*    nsc    = (float*)carve(sizeof(float) * BATCH * PRE_NMS);

    float* o_rois = out;
    float* o_ks   = out + BATCH * POST_NMS * 5;
    float* o_prob = o_ks + BATCH * POST_NMS;
    float* o_bbox = o_prob + BATCH * 18 * NPIX;

    pool_kernel<<<(BATCH * NTGT * 512 + 255) / 256, 256, 0, stream>>>(tgt, pooled);
    wconv_cvt<<<(512 * 1024 * 9 + 255) / 256, 256, 0, stream>>>(Wcc, WccH, 1024);
    wconv_cvt<<<(512 * 1024 * 9 + 255) / 256, 256, 0, stream>>>(Wdf, WdfH, 1024);
    wconv_cvt<<<(512 * 1536 * 9 + 255) / 256, 256, 0, stream>>>(Wc1, Wc1H, 1536);
    wsb_cvt<<<(54 * 512 + 255) / 256, 256, 0, stream>>>(Wsc, Wbb, WsbH);
    imcopy_kernel<<<(BATCH * 512 * NPIX + 255) / 256, 256, 0, stream>>>(im, featH);
    zkey_kernel<<<(BATCH * SORTN + 255) / 256, 256, 0, stream>>>(skey);

    dim3 cgrid(64, 4, BATCH);
    conv3_wmma<<<cgrid, 256, 0, stream>>>(im, featH, pooled, WccH, bcc, featH,
                                          0, 1024, 1536 * NPIX, 0);
    conv3_wmma<<<cgrid, 256, 0, stream>>>(im, featH, pooled, WdfH, bdf, featH,
                                          1, 1024, 1536 * NPIX, 1024);
    conv3_wmma<<<cgrid, 256, 0, stream>>>(im, featH, pooled, Wc1H, bc1, rpnH,
                                          2, 1536, 512 * NPIX, 0);

    head_kernel<<<dim3(NPIX / 256, BATCH), 256, 0, stream>>>(
        rpnH, WsbH, bsc, bbb, o_prob, o_bbox, skey, props, fsc);
    sort_kernel<<<BATCH, 1024, 0, stream>>>(skey);
    nms_kernel<<<BATCH, 1024, 0, stream>>>(skey, props, fsc, nbx, nsc, o_rois, o_ks);

    (void)in_sizes; (void)n_in; (void)out_size; (void)ws_size;
}
